// xFuserMXFP4Linear_63324997812701
// MI455X (gfx1250) — 8-bit matrix operand order repaired and checked by reading, compiled, not run on hardware
//
#include <hip/hip_runtime.h>
#include <stdint.h>


typedef int   v16i __attribute__((ext_vector_type(16)));
typedef float v8f  __attribute__((ext_vector_type(8)));

#define MK_M 8192
#define MK_K 2048
#define MK_N 2048

__device__ __forceinline__ unsigned f32_to_e4m3(float x) {
  unsigned s = (__float_as_uint(x) >> 31) << 7;
  float ax = fabsf(x);
  if (ax != ax) return s | 0x7fu;
  if (ax == 0.0f) return s;
  if (ax >= 448.0f) return s | 0x7eu;
  int e;
  float m = frexpf(ax, &e);
  int ex = e - 1;
  if (ex >= -6) {
    int mant = (int)rintf((m * 2.0f - 1.0f) * 8.0f);
    if (mant == 8) { mant = 0; ++ex; if (ex > 8) return s | 0x7eu; }
    return s | (unsigned)((ex + 7) << 3) | (unsigned)mant;
  }
  int mant = (int)rintf(ldexpf(ax, 9));
  if (mant >= 8) return s | (1u << 3);
  return s | (unsigned)mant;
}

__global__ __launch_bounds__(256)
void mxfp4_quant_to_fp8(const float* __restrict__ x,
                        unsigned char* __restrict__ q8,
                        int nblocks) {
  int b = blockIdx.x * blockDim.x + threadIdx.x;
  if (b >= nblocks) return;

  const float4* src = (const float4*)(x + (size_t)b * 32);
  float v[32];
#pragma unroll
  for (int i = 0; i < 8; ++i) {
    float4 t = src[i];
    v[4*i+0] = t.x; v[4*i+1] = t.y; v[4*i+2] = t.z; v[4*i+3] = t.w;
  }

  float amax = 0.0f;
#pragma unroll
  for (int i = 0; i < 32; ++i) amax = fmaxf(amax, fabsf(v[i]));

  float scale = 1.0f, inv = 1.0f;
  if (amax > 0.0f) {
    int e; (void)frexpf(amax, &e);
    scale = ldexpf(1.0f, e - 3);
    inv   = ldexpf(1.0f, 3 - e);
  }

  unsigned packed[8];
#pragma unroll
  for (int i = 0; i < 8; ++i) {
    unsigned w = 0;
#pragma unroll
    for (int j = 0; j < 4; ++j) {
      float q = v[4*i+j] * inv;
      q = fminf(6.0f, fmaxf(-6.0f, q));
      float aq  = fabsf(q);
      float lsb = aq < 2.0f ? 0.5f : (aq < 4.0f ? 1.0f : 2.0f);
      float qr  = rintf(q / lsb) * lsb;
      w |= f32_to_e4m3(qr * scale) << (8 * j);
    }
    packed[i] = w;
  }

  uint4* dst = (uint4*)(q8 + (size_t)b * 32);
  dst[0] = make_uint4(packed[0], packed[1], packed[2], packed[3]);
  dst[1] = make_uint4(packed[4], packed[5], packed[6], packed[7]);
}

__global__ __launch_bounds__(256)
void mxfp4_gemm_fp8_wmma(const unsigned char* __restrict__ A8,
                         const unsigned char* __restrict__ W8,
                         const float* __restrict__ bias,
                         float* __restrict__ out) {
  const int lane = threadIdx.x & 31;
  const int wave = threadIdx.x >> 5;
  const int wm   = wave & 3;
  const int wn   = wave >> 2;
  const int M0   = blockIdx.y * 128 + wm * 32;
  const int N0   = blockIdx.x * 128 + wn * 64;
  const int l16  = lane & 15;
  const int h    = lane >> 4;

  v8f acc[2][4];
#pragma unroll
  for (int mi = 0; mi < 2; ++mi)
#pragma unroll
    for (int ni = 0; ni < 4; ++ni)
#pragma unroll
      for (int r = 0; r < 8; ++r) acc[mi][ni][r] = 0.0f;

  const unsigned char* aRow[2];
  aRow[0] = A8 + (size_t)(M0 + l16) * MK_K;
  aRow[1] = aRow[0] + 16 * MK_K;
  const unsigned char* bRow[4];
#pragma unroll
  for (int ni = 0; ni < 4; ++ni)
    bRow[ni] = W8 + (size_t)(N0 + ni * 16 + l16) * MK_K;

  for (int k0 = 0; k0 < MK_K; k0 += 128) {
    v16i a[2], b[4];
#pragma unroll
    for (int mi = 0; mi < 2; ++mi) {
#pragma unroll
      for (int j = 0; j < 8; ++j) {
        uint2 t = *(const uint2*)(aRow[mi] + k0 + (j >> 1) * 32 + h * 16 + (j & 1) * 8);
        a[mi][2*j+0] = (int)t.x;
        a[mi][2*j+1] = (int)t.y;
      }
    }
#pragma unroll
    for (int ni = 0; ni < 4; ++ni) {
#pragma unroll
      for (int g = 0; g < 4; ++g) {
        uint4 t = *(const uint4*)(bRow[ni] + k0 + g * 32 + h * 16);
        b[ni][4*g+0] = (int)t.x;
        b[ni][4*g+1] = (int)t.y;
        b[ni][4*g+2] = (int)t.z;
        b[ni][4*g+3] = (int)t.w;
      }
    }
#pragma unroll
    for (int mi = 0; mi < 2; ++mi)
#pragma unroll
      for (int ni = 0; ni < 4; ++ni)
        acc[mi][ni] = __builtin_amdgcn_wmma_f32_16x16x128_fp8_fp8(
            a[mi], b[ni], (short)0, acc[mi][ni], false, false);
  }

#pragma unroll
  for (int mi = 0; mi < 2; ++mi) {
    const int mbase = M0 + mi * 16 + h * 8;
#pragma unroll
    for (int ni = 0; ni < 4; ++ni) {
      const int n = N0 + ni * 16 + l16;
      const float bv = bias[n];
#pragma unroll
      for (int r = 0; r < 8; ++r)
        out[(size_t)(mbase + r) * MK_N + n] = acc[mi][ni][r] + bv;
    }
  }
}

extern "C" void kernel_launch(void* const* d_in, const int* in_sizes, int n_in,
                              void* d_out, int out_size, void* d_ws, size_t ws_size,
                              hipStream_t stream) {
  const float* inp  = (const float*)d_in[0];
  const float* wgt  = (const float*)d_in[1];
  const float* bias = (const float*)d_in[2];
  float* out = (float*)d_out;

  unsigned char* A8 = (unsigned char*)d_ws;
  unsigned char* W8 = A8 + (size_t)MK_M * MK_K;

  const int ablocks = MK_M * MK_K / 32;
  const int wblocks = MK_N * MK_K / 32;
  mxfp4_quant_to_fp8<<<ablocks / 256, 256, 0, stream>>>(inp, A8, ablocks);
  mxfp4_quant_to_fp8<<<wblocks / 256, 256, 0, stream>>>(wgt, W8, wblocks);

  dim3 grid(MK_N / 128, MK_M / 128);
  mxfp4_gemm_fp8_wmma<<<grid, 256, 0, stream>>>(A8, W8, bias, out);
}
